// NonLocalBlock3D_74586402062490
// MI455X (gfx1250) — compile-verified, matrix operand order checked by reading, not run on hardware
//
#include <hip/hip_runtime.h>

typedef __bf16 bf16;
typedef __attribute__((ext_vector_type(16))) __bf16 bf16x16;
typedef __attribute__((ext_vector_type(4)))  __bf16 bf16x4;
typedef __attribute__((ext_vector_type(8)))  float  fx8;

__device__ __forceinline__ int kmap(int e, int hf) {
    // CDNA5 16-bit A/B fragment element->K mapping (CDNA5 ISA 7.12.2)
    return ((e >> 3) << 4) + (hf << 3) + (e & 7);
}

__device__ __forceinline__ fx8 wmma_bf16(bf16x16 a, bf16x16 b, fx8 c) {
    return __builtin_amdgcn_wmma_f32_16x16x32_bf16(false, a, false, b, (short)0, c, false, false);
}

__device__ __forceinline__ fx8 fx8_zero() {
    fx8 z;
#pragma unroll
    for (int e = 0; e < 8; ++e) z[e] = 0.0f;
    return z;
}

// ---------------------------------------------------------------------------
// Kernel W: one-time f32 -> bf16 conversion of Wt|Wp|Wg|Ww into one buffer.
// 131072 elements total; grid (128), block 256, 4 elements/thread.
// ---------------------------------------------------------------------------
__global__ __launch_bounds__(256) void wconv_kernel(
    const float* __restrict__ Wt, const float* __restrict__ Wp,
    const float* __restrict__ Wg, const float* __restrict__ Ww,
    bf16* __restrict__ WB)
{
    const int t = blockIdx.x * 256 + threadIdx.x;   // 0..32767
    const int i4 = t << 2;
    const float* srcs[4] = {Wt, Wp, Wg, Ww};
    const float* src = srcs[i4 >> 15];
    const int off = i4 & 32767;
    float4 v = *reinterpret_cast<const float4*>(src + off);
    bf16x4 o;
    o[0] = (bf16)v.x; o[1] = (bf16)v.y; o[2] = (bf16)v.z; o[3] = (bf16)v.w;
    *reinterpret_cast<bf16x4*>(WB + i4) = o;
}

// ---------------------------------------------------------------------------
// Kernel A: theta/phi/g projections.  x:[B,256,8192] f32, WB: bf16 weights
//   T,P -> [B,8192,128] bf16 (row n major);  G -> [B,128,8192] bf16 (TRANSPOSED)
// grid (64, 2), block 256 (8 waves). Out[n,o] = sum_c W[o,c] x[c,n] + bias[o]
// ---------------------------------------------------------------------------
__global__ __launch_bounds__(256) void proj3_kernel(
    const float* __restrict__ x, const bf16* __restrict__ WB,
    const float* __restrict__ bt, const float* __restrict__ bp, const float* __restrict__ bg,
    bf16* __restrict__ T, bf16* __restrict__ P, bf16* __restrict__ G)
{
    __shared__ bf16 xlds[256][136];   // [c][n] bf16, 16B-aligned pitch
    const int tid = threadIdx.x;
    const int lane = tid & 31, w = tid >> 5;
    const int hf = lane >> 4, ln = lane & 15;
    const int b = blockIdx.y;
    const int n0 = blockIdx.x * 128;

    // stage x tile: rows c=0..255, cols n0..n0+127, f32 -> bf16
    for (int i = tid; i < 256 * 32; i += 256) {
        int r = i >> 5;
        int c4 = (i & 31) << 2;
        float4 v = *reinterpret_cast<const float4*>(
            x + (((size_t)(b * 256 + r)) << 13) + n0 + c4);
        bf16x4 t;
        t[0] = (bf16)v.x; t[1] = (bf16)v.y; t[2] = (bf16)v.z; t[3] = (bf16)v.w;
        *reinterpret_cast<bf16x4*>(&xlds[r][c4]) = t;
    }
    __syncthreads();

    // A-fragments shared by all three projections: A[n][c] = x[c][n]
    bf16x16 aX[8];
#pragma unroll
    for (int kt = 0; kt < 8; ++kt)
#pragma unroll
        for (int e = 0; e < 16; ++e)
            aX[kt][e] = xlds[kt * 32 + kmap(e, hf)][w * 16 + ln];

    const float* Bs[3] = {bt, bp, bg};

#pragma unroll
    for (int s = 0; s < 3; ++s) {
        const bf16* Wm = WB + s * 32768;            // bf16 [128][256]
        fx8 acc[8];
#pragma unroll
        for (int nt = 0; nt < 8; ++nt) acc[nt] = fx8_zero();

        for (int kt = 0; kt < 8; ++kt) {            // K = 256 in chunks of 32
#pragma unroll
            for (int nt = 0; nt < 8; ++nt) {        // N' = o (128)
                bf16x16 bw_;
#pragma unroll
                for (int e = 0; e < 16; ++e)
                    bw_[e] = Wm[(nt * 16 + ln) * 256 + kt * 32 + kmap(e, hf)];
                acc[nt] = wmma_bf16(aX[kt], bw_, acc[nt]);
            }
        }
        const float* bias = Bs[s];
        if (s < 2) {
            bf16* Out = (s == 0) ? T : P;
#pragma unroll
            for (int nt = 0; nt < 8; ++nt) {
                const int o = nt * 16 + ln;
                const float bo = bias[o];
#pragma unroll
                for (int r = 0; r < 8; ++r) {
                    int n = n0 + w * 16 + r + 8 * hf;
                    Out[(((size_t)(b * 8192 + n)) << 7) + o] = (bf16)(acc[nt][r] + bo);
                }
            }
        } else {
            // G: transpose through LDS (reuse xlds) and store [b][o][n] coalesced
            __syncthreads();                         // all waves done reading xlds
            bf16 (*Gbuf)[136] = reinterpret_cast<bf16 (*)[136]>(&xlds[0][0]);
#pragma unroll
            for (int nt = 0; nt < 8; ++nt) {
                const int o = nt * 16 + ln;
                const float bo = bias[o];
#pragma unroll
                for (int r = 0; r < 8; ++r)
                    Gbuf[o][w * 16 + r + 8 * hf] = (bf16)(acc[nt][r] + bo);
            }
            __syncthreads();
            for (int i = tid; i < 128 * 16; i += 256) {
                int o = i >> 4, n8 = (i & 15) << 3;
                *reinterpret_cast<uint4*>(G + (((size_t)(b * 128 + o)) << 13) + n0 + n8) =
                    *reinterpret_cast<const uint4*>(&Gbuf[o][n8]);
            }
        }
    }
}

// ---------------------------------------------------------------------------
// Kernel B: flash attention.  y[n,c] = softmax_m(theta[n,:].phi[:,m]) . g[m,c]
// T,P:[B,8192,128]; G:[B,128,8192] (transposed). grid (64, 2), block 256.
// ---------------------------------------------------------------------------
__global__ __launch_bounds__(256) void flash_kernel(
    const bf16* __restrict__ T, const bf16* __restrict__ P,
    const bf16* __restrict__ G, bf16* __restrict__ Y)
{
    __shared__ bf16  Pt[128][136];     // phi^T tile [m][c]
    __shared__ bf16  Gt[128][136];     // g tile, TRANSPOSED: [c][m]
    __shared__ float Ss[8][16][132];   // per-wave score strip [row][m]
    __shared__ float stt[8][16];       // per-wave per-row scale / inv

    const int tid = threadIdx.x, lane = tid & 31, w = tid >> 5;
    const int hf = lane >> 4, ln = lane & 15;
    const int b = blockIdx.y, n0 = blockIdx.x * 128;

    // persistent theta A-fragments for this wave's 16-row strip
    bf16x16 aT[4];
#pragma unroll
    for (int kt = 0; kt < 4; ++kt)
#pragma unroll
        for (int e = 0; e < 16; ++e)
            aT[kt][e] = T[(((size_t)(b * 8192 + n0 + w * 16 + ln)) << 7) + kt * 32 + kmap(e, hf)];

    fx8 o_acc[8];
#pragma unroll
    for (int ct = 0; ct < 8; ++ct) o_acc[ct] = fx8_zero();
    float m_run = -1e30f, l_run = 0.0f;

    for (int j = 0; j < 64; ++j) {
        const int m0 = j * 128;
        __syncthreads();
        for (int i = tid; i < 128 * 16; i += 256) {
            int r = i >> 4, c8 = (i & 15) << 3;     // 8 bf16 = 16B per op
            *reinterpret_cast<uint4*>(&Pt[r][c8]) =
                *reinterpret_cast<const uint4*>(P + (((size_t)(b * 8192 + m0 + r)) << 7) + c8);
            // G rows are channels c; columns are global positions m
            *reinterpret_cast<uint4*>(&Gt[r][c8]) =
                *reinterpret_cast<const uint4*>(G + (((size_t)(b * 128 + r)) << 13) + m0 + c8);
        }
        __syncthreads();

        // S = theta x phi  (K = Ci = 128)
        fx8 s_acc[8];
#pragma unroll
        for (int nt = 0; nt < 8; ++nt) s_acc[nt] = fx8_zero();
#pragma unroll
        for (int kt = 0; kt < 4; ++kt) {
#pragma unroll
            for (int nt = 0; nt < 8; ++nt) {
                bf16x16 bp_;                         // B[k=c][col=m] = P[m][c]
#pragma unroll
                for (int e = 0; e < 16; ++e)
                    bp_[e] = Pt[nt * 16 + ln][kt * 32 + kmap(e, hf)];
                s_acc[nt] = wmma_bf16(aT[kt], bp_, s_acc[nt]);
            }
        }
        // dump strip to LDS (per-wave buffer; same-wave DS ops are in-order)
#pragma unroll
        for (int nt = 0; nt < 8; ++nt)
#pragma unroll
            for (int r = 0; r < 8; ++r)
                Ss[w][r + 8 * hf][nt * 16 + ln] = s_acc[nt][r];

        // online softmax: 2 lanes per row (halves of 128 cols)
        const int row = ln, c0 = hf * 64;
        float mx = -1e30f;
        for (int c = 0; c < 64; ++c) mx = fmaxf(mx, Ss[w][row][c0 + c]);
        mx = fmaxf(mx, __shfl_xor(mx, 16, 32));
        float m_new = fmaxf(m_run, mx);
        float scale = __expf(m_run - m_new);
        float psum = 0.0f;
        for (int c = 0; c < 64; ++c) {
            float ev = __expf(Ss[w][row][c0 + c] - m_new);
            Ss[w][row][c0 + c] = ev;
            psum += ev;
        }
        l_run = l_run * scale + psum;
        m_run = m_new;
        if (hf == 0) stt[w][row] = scale;

        // rescale running O by per-row factor
        float fr[8];
#pragma unroll
        for (int r = 0; r < 8; ++r) fr[r] = stt[w][r + 8 * hf];
#pragma unroll
        for (int ct = 0; ct < 8; ++ct)
#pragma unroll
            for (int r = 0; r < 8; ++r) o_acc[ct][r] *= fr[r];

        // O += Pexp x G   (K = 128 over m)
#pragma unroll
        for (int kt = 0; kt < 4; ++kt) {
            bf16x16 pa;
#pragma unroll
            for (int e = 0; e < 16; ++e)
                pa[e] = (bf16)Ss[w][ln][kt * 32 + kmap(e, hf)];
#pragma unroll
            for (int ct = 0; ct < 8; ++ct) {
                bf16x16 bg_;                         // B[k=m][col=c] = Gt[c][m], row read
#pragma unroll
                for (int e = 0; e < 16; ++e)
                    bg_[e] = Gt[ct * 16 + ln][kt * 32 + kmap(e, hf)];
                o_acc[ct] = wmma_bf16(pa, bg_, o_acc[ct]);
            }
        }
    }

    // normalize by row sums and store y (bf16)
    float l_tot = l_run + __shfl_xor(l_run, 16, 32);
    float inv = 1.0f / l_tot;
    if (hf == 0) stt[w][ln] = inv;
    float fr[8];
#pragma unroll
    for (int r = 0; r < 8; ++r) fr[r] = stt[w][r + 8 * hf];
#pragma unroll
    for (int ct = 0; ct < 8; ++ct)
#pragma unroll
        for (int r = 0; r < 8; ++r) {
            int n = n0 + w * 16 + r + 8 * hf;
            Y[(((size_t)(b * 8192 + n)) << 7) + ct * 16 + ln] = (bf16)(o_acc[ct][r] * fr[r]);
        }
}

// ---------------------------------------------------------------------------
// Kernel C: output projection + deterministic BN partial stats.
// wy[n,o] = sum_c Ww[o,c] y[n,c] + bw[o];  WY:[B,8192,256] f32
// partial sums -> ps/pq[(b*64+ntile)*256 + o]   (no atomics)
// grid (64, 2, 2) = (ntile, ohalf, b), block 256.
// ---------------------------------------------------------------------------
__global__ __launch_bounds__(256) void outproj_kernel(
    const bf16* __restrict__ Y, const bf16* __restrict__ Wwb, const float* __restrict__ bw,
    float* __restrict__ WY, float* __restrict__ ps, float* __restrict__ pq)
{
    __shared__ bf16  Yt[128][136];
    __shared__ float red_s[8][128];
    __shared__ float red_q[8][128];
    const int tid = threadIdx.x, lane = tid & 31, w = tid >> 5;
    const int hf = lane >> 4, ln = lane & 15;
    const int b = blockIdx.z, oh = blockIdx.y, n0 = blockIdx.x * 128;

    for (int i = tid; i < 128 * 16; i += 256) {
        int r = i >> 4, c8 = (i & 15) << 3;
        *reinterpret_cast<uint4*>(&Yt[r][c8]) =
            *reinterpret_cast<const uint4*>(Y + (((size_t)(b * 8192 + n0 + r)) << 7) + c8);
    }
    __syncthreads();

    fx8 acc[8];
#pragma unroll
    for (int nt = 0; nt < 8; ++nt) acc[nt] = fx8_zero();
#pragma unroll
    for (int kt = 0; kt < 4; ++kt) {
        bf16x16 a;
#pragma unroll
        for (int e = 0; e < 16; ++e)
            a[e] = Yt[w * 16 + ln][kt * 32 + kmap(e, hf)];
#pragma unroll
        for (int nt = 0; nt < 8; ++nt) {
            bf16x16 bw_;
            int o = oh * 128 + nt * 16 + ln;
#pragma unroll
            for (int e = 0; e < 16; ++e)
                bw_[e] = Wwb[o * 128 + kt * 32 + kmap(e, hf)];
            acc[nt] = wmma_bf16(a, bw_, acc[nt]);
        }
    }

#pragma unroll
    for (int nt = 0; nt < 8; ++nt) {
        int o = oh * 128 + nt * 16 + ln;
        float bo = bw[o];
        float s = 0.0f, q = 0.0f;
#pragma unroll
        for (int r = 0; r < 8; ++r) {
            float v = acc[nt][r] + bo;
            int n = n0 + w * 16 + r + 8 * hf;
            WY[(((size_t)(b * 8192 + n)) << 8) + o] = v;
            s += v;
            q += v * v;
        }
        s += __shfl_xor(s, 16, 32);
        q += __shfl_xor(q, 16, 32);
        if (hf == 0) { red_s[w][nt * 16 + ln] = s; red_q[w][nt * 16 + ln] = q; }
    }
    __syncthreads();
    if (tid < 128) {
        float s = 0.0f, q = 0.0f;
#pragma unroll
        for (int ww = 0; ww < 8; ++ww) { s += red_s[ww][tid]; q += red_q[ww][tid]; }
        const int blk = b * 64 + blockIdx.x;
        ps[blk * 256 + oh * 128 + tid] = s;
        pq[blk * 256 + oh * 128 + tid] = q;
    }
}

// ---------------------------------------------------------------------------
// Kernel C2: deterministic fixed-order reduction of BN partials (128 blocks x 256 ch)
// grid (1), block 256.
// ---------------------------------------------------------------------------
__global__ __launch_bounds__(256) void bn_reduce_kernel(
    const float* __restrict__ ps, const float* __restrict__ pq,
    float* __restrict__ gsum, float* __restrict__ gsq)
{
    const int o = threadIdx.x;
    float s = 0.0f, q = 0.0f;
    for (int blk = 0; blk < 128; ++blk) {
        s += ps[blk * 256 + o];
        q += pq[blk * 256 + o];
    }
    gsum[o] = s;
    gsq[o] = q;
}

// ---------------------------------------------------------------------------
// Kernel D: BN finalize + gamma/beta + residual, with LDS transpose
// out[b,o,n] = (wy[b,n,o]-mean)*rsqrt(var+eps)*gamma+beta + x[b,o,n]
// grid (128, 4, 2) = (ntile64, otile64, b), block 256.
// ---------------------------------------------------------------------------
__global__ __launch_bounds__(256) void bn_out_kernel(
    const float* __restrict__ WY, const float* __restrict__ x,
    const float* __restrict__ gsum, const float* __restrict__ gsq,
    const float* __restrict__ gamma, const float* __restrict__ beta,
    float* __restrict__ out)
{
    __shared__ float tile[64][65];
    const int tid = threadIdx.x;
    const int b = blockIdx.z, ob = blockIdx.y * 64, nb = blockIdx.x * 64;

    for (int i = tid; i < 64 * 64; i += 256) {
        int n = i >> 6, o = i & 63;
        tile[n][o] = WY[(((size_t)(b * 8192 + nb + n)) << 8) + ob + o];
    }
    __syncthreads();
    const float invN = 1.0f / 16384.0f;   // B*N samples per channel
    for (int i = tid; i < 64 * 64; i += 256) {
        int o = i >> 6, n = i & 63;
        int og = ob + o;
        float mean = gsum[og] * invN;
        float var  = gsq[og] * invN - mean * mean;
        float inv  = rsqrtf(var + 1e-5f);
        size_t idx = (((size_t)(b * 256 + og)) << 13) + nb + n;
        out[idx] = (tile[n][o] - mean) * inv * gamma[og] + beta[og] + x[idx];
    }
}

// ---------------------------------------------------------------------------
extern "C" void kernel_launch(void* const* d_in, const int* in_sizes, int n_in,
                              void* d_out, int out_size, void* d_ws, size_t ws_size,
                              hipStream_t stream) {
    const float* x     = (const float*)d_in[0];
    const float* Wt    = (const float*)d_in[1];
    const float* bt    = (const float*)d_in[2];
    const float* Wp    = (const float*)d_in[3];
    const float* bp    = (const float*)d_in[4];
    const float* Wg    = (const float*)d_in[5];
    const float* bg    = (const float*)d_in[6];
    const float* Ww    = (const float*)d_in[7];
    const float* bw    = (const float*)d_in[8];
    const float* gamma = (const float*)d_in[9];
    const float* beta  = (const float*)d_in[10];
    float* out = (float*)d_out;

    char* ws = (char*)d_ws;
    bf16*  T    = (bf16*)(ws + 0x0000000);   // [2,8192,128] bf16 = 4 MB
    bf16*  P    = (bf16*)(ws + 0x0400000);   // phi^T [2,8192,128], 4 MB
    bf16*  G    = (bf16*)(ws + 0x0800000);   // g TRANSPOSED [2,128,8192], 4 MB
    bf16*  Y    = (bf16*)(ws + 0x0C00000);   // [2,8192,128], 4 MB
    float* WY   = (float*)(ws + 0x1000000);  // [2,8192,256] f32 = 16 MB
    float* gsum = (float*)(ws + 0x2000000);  // 256 f32
    float* gsq  = (float*)(ws + 0x2000400);  // 256 f32
    float* ps   = (float*)(ws + 0x2010000);  // [128,256] f32 partial sums
    float* pq   = (float*)(ws + 0x2030000);  // [128,256] f32 partial sumsq
    bf16*  WB   = (bf16*)(ws + 0x2040000);   // bf16 weights: Wt|Wp|Wg|Ww, 256 KB

    wconv_kernel    <<<dim3(128),       256, 0, stream>>>(Wt, Wp, Wg, Ww, WB);
    proj3_kernel    <<<dim3(64, 2),     256, 0, stream>>>(x, WB, bt, bp, bg, T, P, G);
    flash_kernel    <<<dim3(64, 2),     256, 0, stream>>>(T, P, G, Y);
    outproj_kernel  <<<dim3(64, 2, 2),  256, 0, stream>>>(Y, WB + 3 * 32768, bw, WY, ps, pq);
    bn_reduce_kernel<<<dim3(1),         256, 0, stream>>>(ps, pq, gsum, gsq);
    bn_out_kernel   <<<dim3(128, 4, 2), 256, 0, stream>>>(WY, x, gsum, gsq, gamma, beta, out);
}
